// DynamicProjectionModule_14637248545388
// MI455X (gfx1250) — matrix operand order repaired and checked by reading, not compiled, not run on hardware
//
#include <hip/hip_runtime.h>
#include <hip/hip_bf16.h>

typedef __attribute__((ext_vector_type(16))) __bf16 v16bf;
typedef __attribute__((ext_vector_type(8)))  __bf16 v8bf;
typedef __attribute__((ext_vector_type(4)))  __bf16 v4bf;
typedef __attribute__((ext_vector_type(8)))  float  v8f;
typedef __attribute__((ext_vector_type(4)))  float  v4f;
typedef __attribute__((ext_vector_type(4)))  unsigned int v4u;
typedef __attribute__((ext_vector_type(8)))  int    v8i_;
typedef __attribute__((ext_vector_type(4)))  int    v4i_;

namespace {
constexpr int kInner   = 1024;   // N_HEADS * HEAD_DIM
constexpr int kHeads   = 16;
constexpr int kHeadDim = 64;
constexpr int kTokens  = 8192;   // B*S == B*N == 8*1024
constexpr int BM = 64;           // block row tile (tokens)
constexpr int BN = 128;          // block col tile (inner)
constexpr int BK = 32;           // k step == bf16 WMMA depth
// LDS tile rows typed as v8bf[5]: 4 data chunks (32 bf16) + 1 pad chunk.
// Row stride 80B (matches TDM pad: every 16 DWORDs insert 4 DWORDs), and
// every fragment read is a 16B-aligned vector load -> ds_load_b128.
constexpr int RCH = 5;
}

#if __has_builtin(__builtin_amdgcn_tensor_load_to_lds) && \
    __has_builtin(__builtin_amdgcn_s_wait_tensorcnt)
#define HAVE_TDM 1
#else
#define HAVE_TDM 0
#endif

// ---------------- f32 -> bf16 conversion (one-time, NT reads) ---------------
__global__ __launch_bounds__(256)
void cvt_f32_bf16_kernel(const float* __restrict__ in, __bf16* __restrict__ outp, int n4) {
  const int i = blockIdx.x * blockDim.x + threadIdx.x;
  if (i < n4) {
    const v4f f = __builtin_nontemporal_load((const v4f*)in + i);
    v4bf o = { (__bf16)f.x, (__bf16)f.y, (__bf16)f.z, (__bf16)f.w };
    *((v4bf*)outp + i) = o;   // regular store: bf16 copy stays hot in 192MB L2
  }
}

// ---------------- routing: idx[t] = argmax_p dot(x[t,:], Wr[p,:]), P == 3 ----
__global__ __launch_bounds__(256)
void route_argmax_kernel(const float* __restrict__ x, const float* __restrict__ Wr,
                         int* __restrict__ idx, int D) {
  const int wave = (blockIdx.x * blockDim.x + threadIdx.x) >> 5;  // one wave32 / token
  const int lane = threadIdx.x & 31;
  const float* xp = x + (size_t)wave * D;
  float s0 = 0.f, s1 = 0.f, s2 = 0.f;
  for (int d = lane; d < D; d += 32) {
    const float xv = xp[d];
    s0 += xv * Wr[d];
    s1 += xv * Wr[D + d];
    s2 += xv * Wr[2 * D + d];
  }
#pragma unroll
  for (int off = 16; off > 0; off >>= 1) {
    s0 += __shfl_down(s0, off, 32);
    s1 += __shfl_down(s1, off, 32);
    s2 += __shfl_down(s2, off, 32);
  }
  if (lane == 0) {
    int best = 0; float bv = s0;
    if (s1 > bv) { bv = s1; best = 1; }   // strict '>' == jnp.argmax first-max
    if (s2 > bv) { bv = s2; best = 2; }
    idx[wave] = best;
  }
}

// ---------------- TDM 2D tile load: bf16, HW-padded rows --------------------
#if HAVE_TDM
// D# group0/group1 per CDNA5 ISA ch.8. data_size=1 (2B). pad: every 16 DWORDs
// (=64B row) insert 4 DWORDs (16B)  ->  LDS row stride 80B == RCH v8bf chunks.
__device__ __forceinline__ void tdm_load_tile_bf16(unsigned lds_byte_off,
                                                   const __bf16* g,
                                                   unsigned rows, unsigned cols,
                                                   unsigned long long stride_elems) {
  const unsigned long long ga = (unsigned long long)(size_t)g;
  v4u g0;
  g0.x = 1u;                                             // count=1, user descriptor
  g0.y = lds_byte_off;                                   // lds_addr
  g0.z = (unsigned)ga;                                   // global_addr[31:0]
  g0.w = (unsigned)((ga >> 32) & 0x01FFFFFFull) | (2u << 30);  // addr[56:32], type=2
  v8i_ g1;
  g1[0] = (int)((1u << 16) | (1u << 20) | (3u << 22) | (3u << 25)); // 2B, pad_en, 16dw->+4dw
  g1[1] = (int)((cols & 0xFFFFu) << 16);                               // tensor_dim0[15:0]
  g1[2] = (int)(((cols >> 16) & 0xFFFFu) | ((rows & 0xFFFFu) << 16));  // td0[31:16], td1[15:0]
  g1[3] = (int)(((rows >> 16) & 0xFFFFu) | ((cols & 0xFFFFu) << 16));  // td1[31:16], tile_dim0
  g1[4] = (int)(rows & 0xFFFFu);                                       // tile_dim1, tile_dim2=0
  g1[5] = (int)(stride_elems & 0xFFFFFFFFull);                         // dim0_stride[31:0]
  g1[6] = (int)((stride_elems >> 32) & 0xFFFFull);                     // dim0_stride[47:32]
  g1[7] = 0;
  v4i_ z4 = {0, 0, 0, 0};                                // groups 2/3 unused (2D tensor)
  v8i_ z8 = {0, 0, 0, 0, 0, 0, 0, 0};
  __builtin_amdgcn_tensor_load_to_lds(g0, g1, z4, z4, z8, 0);
}
#endif

// ---------------- bf16 WMMA projection: out = to_heads(X @ W[bank]^T) -------
template <int BANKS>
struct SMem {
  v8bf A[2][BM][RCH];          // 80B rows, 16B-aligned chunks
  v8bf Bw[2][BANKS][BN][RCH];
  int  R[BM];
};

template <int BANKS>
__global__ __launch_bounds__(256)
void proj_wmma_kernel(const __bf16* __restrict__ X,   // [kTokens, D]
                      const __bf16* __restrict__ W,   // [BANKS, kInner, D]
                      const int*    __restrict__ route,
                      float* __restrict__ out,        // [B, kHeads, 1024, kHeadDim]
                      int D) {
  __shared__ SMem<BANKS> sm;

  const int tilesN = kInner / BN;                 // 8
  const int row0 = (blockIdx.x / tilesN) * BM;
  const int col0 = (blockIdx.x % tilesN) * BN;

  const int tid  = threadIdx.x;
  const int wave = tid >> 5;
  const int lane = tid & 31;
  const int wm   = (wave & 3) * 16;
  const int wn   = (wave >> 2) * 64;
  const int l16  = lane & 15;
  const int half = lane >> 4;

  if (BANKS > 1 && tid < BM) sm.R[tid] = route[row0 + tid];

  v8f acc[BANKS][4] = {};
  const int nK = D / BK;

#if HAVE_TDM
  // prologue: wave 0 DMAs the first tile set into buffer 0 (TDM issues once per
  // executing wave regardless of EXEC -> restrict to wave 0)
  if (wave == 0) {
    const unsigned oa = (unsigned)((const char*)&sm.A[0][0][0] - (const char*)&sm);
    tdm_load_tile_bf16(oa, X + (size_t)row0 * D, BM, BK, (unsigned long long)D);
#pragma unroll
    for (int b = 0; b < BANKS; ++b) {
      const unsigned ob = (unsigned)((const char*)&sm.Bw[0][b][0][0] - (const char*)&sm);
      tdm_load_tile_bf16(ob, W + ((size_t)b * kInner + col0) * D, BN, BK,
                         (unsigned long long)D);
    }
  }
#endif

  for (int kk = 0; kk < nK; ++kk) {
#if HAVE_TDM
    const int buf = kk & 1;
    if (kk + 1 < nK) {
      if (wave == 0) {   // issue next tile set, then drain the current one
        const int k1 = (kk + 1) * BK;
        const unsigned oa = (unsigned)((const char*)&sm.A[buf ^ 1][0][0] - (const char*)&sm);
        tdm_load_tile_bf16(oa, X + (size_t)row0 * D + k1, BM, BK, (unsigned long long)D);
#pragma unroll
        for (int b = 0; b < BANKS; ++b) {
          const unsigned ob =
              (unsigned)((const char*)&sm.Bw[buf ^ 1][b][0][0] - (const char*)&sm);
          tdm_load_tile_bf16(ob, W + ((size_t)b * kInner + col0) * D + k1, BN, BK,
                             (unsigned long long)D);
        }
        __builtin_amdgcn_s_wait_tensorcnt(BANKS + 1);  // oldest batch (cur buf) done
      }
    } else {
      if (wave == 0) __builtin_amdgcn_s_wait_tensorcnt(0);
    }
    __syncthreads();
#else
    const int buf = 0;
    __syncthreads();
    // synchronous fallback: b128 copies of bf16 tiles
    for (int c = tid; c < BM * (BK / 8); c += 256) {
      const int r = c >> 2, s = c & 3;
      sm.A[0][r][s] = *(const v8bf*)(X + (size_t)(row0 + r) * D + (size_t)kk * BK + s * 8);
    }
#pragma unroll
    for (int b = 0; b < BANKS; ++b)
      for (int c = tid; c < BN * (BK / 8); c += 256) {
        const int r = c >> 2, s = c & 3;
        sm.Bw[0][b][r][s] =
            *(const v8bf*)(W + ((size_t)b * kInner + col0 + r) * D + (size_t)kk * BK + s * 8);
      }
    __syncthreads();
#endif

    // A fragment (16-bit A 16x32 layout): row m = wm + l16;
    // chunk[half]   -> k =  8*half + 0..7   (elems j=0..7)
    // chunk[2+half] -> k = 16 + 8*half + 0..7 (elems j=8..15)
    const int am = wm + l16;
    const v8bf a0 = sm.A[buf][am][half];
    const v8bf a1 = sm.A[buf][am][2 + half];
    const v16bf afrag = __builtin_shufflevector(
        a0, a1, 0, 1, 2, 3, 4, 5, 6, 7, 8, 9, 10, 11, 12, 13, 14, 15);

#pragma unroll
    for (int t = 0; t < 4; ++t) {
      const int bn_ = wn + t * 16 + l16;
#pragma unroll
      for (int b = 0; b < BANKS; ++b) {
        // B fragment (32x16): col n = bn_; chunks [half], [2+half] -> k = 8*half + 0..7, 16 + 8*half + 0..7
        const v8bf b0 = sm.Bw[buf][b][bn_][half];
        const v8bf b1 = sm.Bw[buf][b][bn_][2 + half];
        const v16bf bfrag = __builtin_shufflevector(
            b0, b1, 0, 1, 2, 3, 4, 5, 6, 7, 8, 9, 10, 11, 12, 13, 14, 15);
        acc[b][t] = __builtin_amdgcn_wmma_f32_16x16x32_bf16(
            false, afrag, false, bfrag, (short)0, acc[b][t], false, false);
      }
    }
#if HAVE_TDM
    __syncthreads();  // all reads of `buf` done before it is re-filled next iter
#endif
  }
#if !HAVE_TDM
  __syncthreads();
#endif

  // epilogue. C layout: VGPR v, lanes 0-15 -> (M=v,N=lane); 16-31 -> (M=v+8,N=lane-16)
#pragma unroll
  for (int t = 0; t < 4; ++t) {
    const int ncol = col0 + wn + t * 16 + l16;
    const int h = ncol >> 6, dd = ncol & 63;
#pragma unroll
    for (int v = 0; v < 8; ++v) {
      const int mloc = wm + v + 8 * half;
      const int grow = row0 + mloc;
      float val;
      if constexpr (BANKS == 1) {
        val = acc[0][t][v];
      } else {
        const int sel = sm.R[mloc];
        const float c0 = acc[0][t][v], c1 = acc[1][t][v], c2 = acc[2][t][v];
        val = (sel == 0) ? c0 : ((sel == 1) ? c1 : c2);
      }
      const int bidx = grow >> 10;
      const int ltok = grow & 1023;
      __builtin_nontemporal_store(
          val, &out[(((size_t)bidx * kHeads + h) * 1024 + ltok) * kHeadDim + dd]);
    }
  }
}

extern "C" void kernel_launch(void* const* d_in, const int* in_sizes, int n_in,
                              void* d_out, int out_size, void* d_ws, size_t ws_size,
                              hipStream_t stream) {
  (void)in_sizes; (void)n_in; (void)out_size; (void)ws_size;

  const float* img = (const float*)d_in[0];   // [8,1024,1024]
  const float* txt = (const float*)d_in[1];   // [8,1024,768]
  const float* Wq  = (const float*)d_in[2];   // [1024,1024]
  const float* Wk  = (const float*)d_in[3];   // [1024,768]
  const float* Wv  = (const float*)d_in[4];   // [1024,768]
  const float* Wqb = (const float*)d_in[5];   // [3,1024,1024]
  const float* Wkb = (const float*)d_in[6];   // [3,1024,768]
  const float* Wrq = (const float*)d_in[7];   // [3,1024]
  const float* Wrk = (const float*)d_in[8];   // [3,768]

  // ---- workspace layout ----
  char* ws = (char*)d_ws;
  int* q_idx = (int*)ws;                 ws += (size_t)kTokens * 4;
  int* k_idx = (int*)ws;                 ws += (size_t)kTokens * 4;
  const size_t nImg = (size_t)8 * 1024 * 1024, nTxt = (size_t)8 * 1024 * 768;
  const size_t nWq = (size_t)1024 * 1024, nWkv = (size_t)1024 * 768;
  __bf16* img_bf = (__bf16*)ws;  ws += nImg * 2;
  __bf16* txt_bf = (__bf16*)ws;  ws += nTxt * 2;
  __bf16* Wq_bf  = (__bf16*)ws;  ws += nWq * 2;
  __bf16* Wk_bf  = (__bf16*)ws;  ws += nWkv * 2;
  __bf16* Wv_bf  = (__bf16*)ws;  ws += nWkv * 2;
  __bf16* Wqb_bf = (__bf16*)ws;  ws += 3 * nWq * 2;
  __bf16* Wkb_bf = (__bf16*)ws;  ws += 3 * nWkv * 2;

  // routing (reads original f32; one wave per token)
  route_argmax_kernel<<<kTokens / 8, 256, 0, stream>>>(img, Wrq, q_idx, 1024);
  route_argmax_kernel<<<kTokens / 8, 256, 0, stream>>>(txt, Wrk, k_idx, 768);

  // one-time bf16 conversion (all sizes divisible by 1024)
  auto cvt = [&](const float* src, __bf16* dst, size_t n) {
    const int n4 = (int)(n / 4);
    cvt_f32_bf16_kernel<<<(n4 + 255) / 256, 256, 0, stream>>>(src, dst, n4);
  };
  cvt(img, img_bf, nImg);
  cvt(txt, txt_bf, nTxt);
  cvt(Wq,  Wq_bf,  nWq);
  cvt(Wk,  Wk_bf,  nWkv);
  cvt(Wv,  Wv_bf,  nWkv);
  cvt(Wqb, Wqb_bf, 3 * nWq);
  cvt(Wkb, Wkb_bf, 3 * nWkv);

  float* outQ  = (float*)d_out;
  const size_t one = (size_t)8 * kHeads * 1024 * kHeadDim;
  float* outK  = outQ  + one;
  float* outV  = outK  + one;
  float* outQp = outV  + one;
  float* outKp = outQp + one;

  const dim3 grid((kTokens / BM) * (kInner / BN));   // 1024 blocks
  proj_wmma_kernel<1><<<grid, 256, 0, stream>>>(img_bf, Wq_bf,  nullptr, outQ,  1024);
  proj_wmma_kernel<1><<<grid, 256, 0, stream>>>(txt_bf, Wk_bf,  nullptr, outK,  768);
  proj_wmma_kernel<1><<<grid, 256, 0, stream>>>(txt_bf, Wv_bf,  nullptr, outV,  768);
  proj_wmma_kernel<3><<<grid, 256, 0, stream>>>(img_bf, Wqb_bf, q_idx,   outQp, 1024);
  proj_wmma_kernel<3><<<grid, 256, 0, stream>>>(txt_bf, Wkb_bf, k_idx,   outKp, 768);
}
